// SingleStreamBlock_31782757990867
// MI455X (gfx1250) — matrix operand order repaired and checked by reading, not compiled, not run on hardware
//
#include <hip/hip_runtime.h>
#include <hip/hip_bf16.h>
#include <stdint.h>

// Problem constants (B=1)
#define L_    2048
#define H_    3072
#define NH_   24
#define HD_   128
#define MLP_  12288
#define OUT1_ (3*H_ + MLP_)   // 21504
#define CAT_  (H_ + MLP_)     // 15360
#define EPS_  1e-6f

typedef __attribute__((ext_vector_type(16))) __bf16 v16bf;
typedef __attribute__((ext_vector_type(8)))  float  v8f;

union BFrag {
  v16bf v;
  uint4 q[2];
  unsigned short u[16];
};

static __device__ __forceinline__ unsigned short f2bf(float f) {
  unsigned int u = __builtin_bit_cast(unsigned int, f);
  u += 0x7fffu + ((u >> 16) & 1u);            // round-to-nearest-even
  return (unsigned short)(u >> 16);
}
static __device__ __forceinline__ float bf2f(unsigned short h) {
  unsigned int u = ((unsigned int)h) << 16;
  return __builtin_bit_cast(float, u);
}

// CDNA5 async memory->LDS copy (ASYNCcnt-tracked). offset: applies to both
// the LDS destination and the global source per CDNA5 ISA.
template <int OFF>
static __device__ __forceinline__ void async_b128(unsigned int lds_off,
                                                  const void* gaddr) {
  asm volatile("global_load_async_to_lds_b128 %0, %1, off offset:%2"
               :: "v"(lds_off), "v"(gaddr), "n"(OFF) : "memory");
}
static __device__ __forceinline__ void wait_async0() {
  asm volatile("s_wait_asynccnt 0x0" ::: "memory");
}
static __device__ __forceinline__ unsigned int lds_off32(const void* p) {
  // flat shared address: low 32 bits are the wave-relative LDS byte offset
  return (unsigned int)(uintptr_t)p;
}

// ---------------------------------------------------------------------------
// 1) mod = silu(vec) @ mod_w.T + mod_b        (9216 rows, K=3072)
// ---------------------------------------------------------------------------
__global__ void mod_gemv_kernel(const float* __restrict__ vec,
                                const float* __restrict__ w,
                                const float* __restrict__ b,
                                float* __restrict__ mod) {
  int row = blockIdx.x;
  int tid = threadIdx.x;                       // 128 threads
  const float* wr = w + (size_t)row * H_;
  float acc = 0.f;
  for (int k = tid; k < H_; k += 128) {
    float v = vec[k];
    float s = v / (1.f + __expf(-v));
    acc += s * wr[k];
  }
  for (int m = 16; m >= 1; m >>= 1) acc += __shfl_xor(acc, m, 32);
  __shared__ float red[4];
  int w4 = tid >> 5, ln = tid & 31;
  if (ln == 0) red[w4] = acc;
  __syncthreads();
  if (tid == 0) mod[row] = red[0] + red[1] + red[2] + red[3] + b[row];
}

// ---------------------------------------------------------------------------
// 2) LayerNorm + modulate -> bf16 x_mod       (one block per row)
// ---------------------------------------------------------------------------
__global__ void ln_mod_kernel(const float* __restrict__ x,
                              const float* __restrict__ mod,
                              unsigned short* __restrict__ xm) {
  int row = blockIdx.x;
  int tid = threadIdx.x;                       // 256 threads
  const float* xr = x + (size_t)row * H_;
  float xv[12];
  float s = 0.f;
#pragma unroll
  for (int i = 0; i < 12; i++) { xv[i] = xr[tid + i*256]; s += xv[i]; }
  for (int m = 16; m >= 1; m >>= 1) s += __shfl_xor(s, m, 32);
  __shared__ float red[8];
  int w = tid >> 5, ln = tid & 31;
  if (ln == 0) red[w] = s;
  __syncthreads();
  float tot = 0.f;
#pragma unroll
  for (int i = 0; i < 8; i++) tot += red[i];
  float mu = tot * (1.f / H_);
  __syncthreads();
  float s2 = 0.f;
#pragma unroll
  for (int i = 0; i < 12; i++) { float d = xv[i] - mu; s2 += d*d; }
  for (int m = 16; m >= 1; m >>= 1) s2 += __shfl_xor(s2, m, 32);
  if (ln == 0) red[w] = s2;
  __syncthreads();
  float tot2 = 0.f;
#pragma unroll
  for (int i = 0; i < 8; i++) tot2 += red[i];
  float rinv = rsqrtf(tot2 * (1.f / H_) + EPS_);
#pragma unroll
  for (int i = 0; i < 12; i++) {
    int c = tid + i*256;
    float v = (1.f + mod[H_ + c]) * ((xv[i] - mu) * rinv) + mod[c];
    xm[(size_t)row * H_ + c] = f2bf(v);
  }
}

// ---------------------------------------------------------------------------
// 3) fp32 -> bf16 converter (weights)
// ---------------------------------------------------------------------------
__global__ void cvt_bf16_kernel(const float* __restrict__ src,
                                unsigned short* __restrict__ dst, long n) {
  long i = (long)blockIdx.x * blockDim.x + threadIdx.x;
  long stride = (long)gridDim.x * blockDim.x;
  for (; i < n; i += stride) dst[i] = f2bf(src[i]);
}

// ---------------------------------------------------------------------------
// 4) Tiled WMMA GEMM:  C[M,N] = A[M,K] @ W[N,K]^T + bias
//    Block tile 128x256, 8 waves (2x4), wave tile 64x64, K-step 32.
//    Double-buffered LDS filled by global_load_async_to_lds_b128;
//    one barrier + one s_wait_asynccnt per K-step.
//    EPI==0: store bf16 (h).  EPI==1: y = clip(x + gate*(C+bias)) f32.
// ---------------------------------------------------------------------------
template<int EPI>
__global__ __launch_bounds__(256) void gemm_bf16_wmma(
    const unsigned short* __restrict__ A,     // [M,K] bf16
    const unsigned short* __restrict__ W,     // [N,K] bf16
    const float* __restrict__ bias,           // [N]
    int M, int N, int K,
    unsigned short* __restrict__ Hout,        // EPI 0
    const float* __restrict__ X,              // EPI 1 (residual)
    const float* __restrict__ mod,            // EPI 1 (gate at mod[2H+n])
    float* __restrict__ Yout) {               // EPI 1
  __shared__ unsigned short Alds[2][128 * 32];   // 2 x 8 KB
  __shared__ unsigned short Blds[2][256 * 32];   // 2 x 16 KB
  int m0 = blockIdx.y * 128, n0 = blockIdx.x * 256;
  int tid = threadIdx.x, wave = tid >> 5, lane = tid & 31;
  int mwb = (wave >> 2) * 64, nwb = (wave & 3) * 64;
  int hi = lane >> 4, nn = lane & 15, g8 = hi * 8;

  v8f vz = {0.f,0.f,0.f,0.f,0.f,0.f,0.f,0.f};
  v8f acc[4][4];
#pragma unroll
  for (int i = 0; i < 4; i++)
#pragma unroll
    for (int j = 0; j < 4; j++) acc[i][j] = vz;

  // Async fill mapping: A tile 128x32 (32B/thread), B tile 256x32 (64B/thread)
  int lrow = tid >> 1;                // 0..127
  int lcol = (tid & 1) * 16;          // 0 / 16
  const unsigned short* gA = A + (size_t)(m0 + lrow) * K + lcol;
  const unsigned short* gB = W + (size_t)(n0 + tid) * K;
  unsigned int ldsA0 = lds_off32(&Alds[0][lrow*32 + lcol]);
  unsigned int ldsA1 = lds_off32(&Alds[1][lrow*32 + lcol]);
  unsigned int ldsB0 = lds_off32(&Blds[0][tid*32]);
  unsigned int ldsB1 = lds_off32(&Blds[1][tid*32]);

  auto issue_tile = [&](int buf, int k0) {
    unsigned int la = buf ? ldsA1 : ldsA0;
    unsigned int lb = buf ? ldsB1 : ldsB0;
    const unsigned short* ga = gA + k0;
    const unsigned short* gb = gB + k0;
    async_b128<0>(la, ga);
    async_b128<16>(la, ga);
    async_b128<0>(lb, gb);
    async_b128<16>(lb, gb);
    async_b128<32>(lb, gb);
    async_b128<48>(lb, gb);
  };

  issue_tile(0, 0);
  int p = 0;
  for (int k0 = 0; k0 < K; k0 += 32) {
    wait_async0();              // my fills of buf p are complete
    __syncthreads();            // everyone's fills done; prev reads of buf 1-p done
    if (k0 + 32 < K) issue_tile(1 - p, k0 + 32);

    const unsigned short* Ab = &Alds[p][0];
    const unsigned short* Bb = &Blds[p][0];
    BFrag af[4];
#pragma unroll
    for (int ms = 0; ms < 4; ms++) {
      int r = mwb + ms*16 + nn;         // A row; lane half selects K group
      af[ms].q[0] = *(const uint4*)&Ab[r*32 + g8];
      af[ms].q[1] = *(const uint4*)&Ab[r*32 + 16 + g8];
    }
#pragma unroll
    for (int ns = 0; ns < 4; ns++) {
      BFrag bf;
      int c = nwb + ns*16 + nn;         // B column = W row; lane half selects K group
      bf.q[0] = *(const uint4*)&Bb[c*32 + g8];
      bf.q[1] = *(const uint4*)&Bb[c*32 + 16 + g8];
#pragma unroll
      for (int ms = 0; ms < 4; ms++)
        acc[ms][ns] = __builtin_amdgcn_wmma_f32_16x16x32_bf16(
            false, af[ms].v, false, bf.v, (short)0, acc[ms][ns], false, false);
    }
    p ^= 1;
  }

#pragma unroll
  for (int ms = 0; ms < 4; ms++)
#pragma unroll
    for (int ns = 0; ns < 4; ns++)
#pragma unroll
      for (int r = 0; r < 8; r++) {
        int gm = m0 + mwb + ms*16 + r + 8*hi;
        int gn = n0 + nwb + ns*16 + nn;
        float v = acc[ms][ns][r] + bias[gn];
        if (EPI == 0) {
          Hout[(size_t)gm * N + gn] = f2bf(v);
        } else {
          float y = X[(size_t)gm * H_ + gn] + mod[2*H_ + gn] * v;
          y = fminf(fmaxf(y, -65504.f), 65504.f);
          Yout[(size_t)gm * H_ + gn] = y;
        }
      }
}

// ---------------------------------------------------------------------------
// 5) RMS-norm + RoPE + repack q/k/v head-major     (one block per (l,head))
// ---------------------------------------------------------------------------
__global__ void qkv_post_kernel(const unsigned short* __restrict__ h,
                                const float* __restrict__ pe,
                                const float* __restrict__ qs,
                                const float* __restrict__ ks,
                                unsigned short* __restrict__ qb,
                                unsigned short* __restrict__ kb,
                                unsigned short* __restrict__ vb) {
  int blk = blockIdx.x;
  int head = blk % NH_;
  int l    = blk / NH_;
  int d = threadIdx.x;                         // 128 threads
  size_t hbase = (size_t)l * OUT1_ + head * HD_;
  float q = bf2f(h[hbase + d]);
  float k = bf2f(h[hbase + H_ + d]);
  unsigned short vraw = h[hbase + 2*H_ + d];
  float sq = q*q, sk = k*k;
  for (int m = 16; m >= 1; m >>= 1) { sq += __shfl_xor(sq, m, 32); sk += __shfl_xor(sk, m, 32); }
  __shared__ float red[2][4];
  int w = d >> 5, ln = d & 31;
  if (ln == 0) { red[0][w] = sq; red[1][w] = sk; }
  __syncthreads();
  float tq = red[0][0] + red[0][1] + red[0][2] + red[0][3];
  float tk = red[1][0] + red[1][1] + red[1][2] + red[1][3];
  float qn = q * rsqrtf(tq * (1.f/HD_) + EPS_) * qs[d];
  float kn = k * rsqrtf(tk * (1.f/HD_) + EPS_) * ks[d];
  // RoPE: pe is (L, 64, 2, 2); pair partner via shfl
  int a = d & 1;
  const float* p = pe + (size_t)l * 256 + (d >> 1) * 4 + a * 2;
  float qo = __shfl_xor(qn, 1, 32);
  float ko = __shfl_xor(kn, 1, 32);
  float q0 = a ? qo : qn, q1 = a ? qn : qo;
  float k0 = a ? ko : kn, k1 = a ? kn : ko;
  float qr = p[0]*q0 + p[1]*q1;
  float kr = p[0]*k0 + p[1]*k1;
  size_t obase = ((size_t)head * L_ + l) * HD_ + d;
  qb[obase] = f2bf(qr);
  kb[obase] = f2bf(kr);
  vb[obase] = vraw;
}

// ---------------------------------------------------------------------------
// 6) GELU(tanh) of mlp part -> concat buffer (cols H..CAT)
// ---------------------------------------------------------------------------
__global__ void gelu_kernel(const unsigned short* __restrict__ h,
                            unsigned short* __restrict__ cat) {
  long n = (long)L_ * MLP_;
  long i = (long)blockIdx.x * blockDim.x + threadIdx.x;
  long stride = (long)gridDim.x * blockDim.x;
  for (; i < n; i += stride) {
    long l = i / MLP_, j = i - l * MLP_;
    float x = bf2f(h[l * OUT1_ + 3*H_ + j]);
    float t = tanhf(0.7978845608028654f * (x + 0.044715f * x * x * x));
    float g = 0.5f * x * (1.f + t);
    cat[l * CAT_ + H_ + j] = f2bf(g);
  }
}

// ---------------------------------------------------------------------------
// 7) Flash attention: per (head, 128-row q tile); 32-key KV tiles via LDS.
//    8 waves; wave owns 16 q rows. S=Q@K^T (8 wmma), online softmax,
//    O+=P@V (8 wmma). o written into concat buffer cols 0..H.
// ---------------------------------------------------------------------------
__global__ __launch_bounds__(256) void flash_attn_kernel(
    const unsigned short* __restrict__ qb,
    const unsigned short* __restrict__ kb,
    const unsigned short* __restrict__ vb,
    unsigned short* __restrict__ cat) {
  __shared__ unsigned short Kls[32 * 128];     // [key][hd]
  __shared__ unsigned short Vls[128 * 32];     // [hd][key]  (transposed)
  __shared__ unsigned short Pls[8][16 * 32];   // per-wave P tile
  const int nqt = L_ / 128;                    // 16
  int head = blockIdx.x / nqt;
  int qt   = blockIdx.x % nqt;
  int tid = threadIdx.x, wave = tid >> 5, lane = tid & 31;
  int hi = lane >> 4, nn = lane & 15, g8 = hi * 8;

  // Q fragments, resident whole kernel (4 K-steps of HD=128)
  int qrow = qt*128 + wave*16 + nn;
  const unsigned short* qp = qb + ((size_t)head * L_ + qrow) * HD_;
  BFrag qf[4];
#pragma unroll
  for (int kk = 0; kk < 4; kk++) {
    qf[kk].q[0] = *(const uint4*)(qp + kk*32 + g8);
    qf[kk].q[1] = *(const uint4*)(qp + kk*32 + 16 + g8);
  }

  v8f vz = {0.f,0.f,0.f,0.f,0.f,0.f,0.f,0.f};
  v8f Oacc[8];
  float mrun[8], lrun[8];
#pragma unroll
  for (int i = 0; i < 8; i++) { Oacc[i] = vz; mrun[i] = -3.0e38f; lrun[i] = 0.f; }
  const float sm = 0.08838834764831845f;       // 1/sqrt(128)

  int lr = tid >> 3;                           // 0..31 (key row)
  int lc = (tid & 7) * 16;                     // 0..112 (hd segment)
  unsigned short* Pw = &Pls[wave][0];

  for (int kv0 = 0; kv0 < L_; kv0 += 32) {
    const unsigned short* kp = kb + ((size_t)head * L_ + kv0 + lr) * HD_ + lc;
    const unsigned short* vp = vb + ((size_t)head * L_ + kv0 + lr) * HD_ + lc;
    *(uint4*)&Kls[lr*128 + lc]     = *(const uint4*)kp;
    *(uint4*)&Kls[lr*128 + lc + 8] = *(const uint4*)(kp + 8);
    BFrag vt;
    vt.q[0] = *(const uint4*)vp;
    vt.q[1] = *(const uint4*)(vp + 8);
#pragma unroll
    for (int j = 0; j < 16; j++) Vls[(lc + j)*32 + lr] = vt.u[j];
    __syncthreads();

    // S = Q @ K^T  (two 16-key subtiles)
    v8f s0 = vz, s1 = vz;
#pragma unroll
    for (int kk = 0; kk < 4; kk++) {
      BFrag b0, b1;
      b0.q[0] = *(const uint4*)&Kls[nn*128 + kk*32 + g8];
      b0.q[1] = *(const uint4*)&Kls[nn*128 + kk*32 + 16 + g8];
      b1.q[0] = *(const uint4*)&Kls[(16+nn)*128 + kk*32 + g8];
      b1.q[1] = *(const uint4*)&Kls[(16+nn)*128 + kk*32 + 16 + g8];
      s0 = __builtin_amdgcn_wmma_f32_16x16x32_bf16(false, qf[kk].v, false, b0.v, (short)0, s0, false, false);
      s1 = __builtin_amdgcn_wmma_f32_16x16x32_bf16(false, qf[kk].v, false, b1.v, (short)0, s1, false, false);
    }

    // Online softmax per tracked row r (actual row = r + 8*hi)
#pragma unroll
    for (int r = 0; r < 8; r++) {
      float a0 = s0[r] * sm, a1 = s1[r] * sm;
      float mx = fmaxf(a0, a1);
      for (int m = 8; m >= 1; m >>= 1) mx = fmaxf(mx, __shfl_xor(mx, m, 32));
      float mnew = fmaxf(mrun[r], mx);
      float corr = __expf(mrun[r] - mnew);
      float p0 = __expf(a0 - mnew), p1 = __expf(a1 - mnew);
      float rs = p0 + p1;
      for (int m = 8; m >= 1; m >>= 1) rs += __shfl_xor(rs, m, 32);
      lrun[r] = lrun[r] * corr + rs;
      mrun[r] = mnew;
#pragma unroll
      for (int nt = 0; nt < 8; nt++) Oacc[nt][r] *= corr;
      int mr = r + 8*hi;
      Pw[mr*32 + nn]      = f2bf(p0);
      Pw[mr*32 + 16 + nn] = f2bf(p1);
    }

    // Re-fragment P as WMMA A operand (same-wave LDS round trip)
    BFrag pf;
    pf.q[0] = *(const uint4*)&Pw[nn*32 + g8];
    pf.q[1] = *(const uint4*)&Pw[nn*32 + 16 + g8];

    // O += P @ V
#pragma unroll
    for (int nt = 0; nt < 8; nt++) {
      BFrag vf;
      int c = nt*16 + nn;
      vf.q[0] = *(const uint4*)&Vls[c*32 + g8];
      vf.q[1] = *(const uint4*)&Vls[c*32 + 16 + g8];
      Oacc[nt] = __builtin_amdgcn_wmma_f32_16x16x32_bf16(false, pf.v, false, vf.v, (short)0, Oacc[nt], false, false);
    }
    __syncthreads();
  }

#pragma unroll
  for (int nt = 0; nt < 8; nt++)
#pragma unroll
    for (int r = 0; r < 8; r++) {
      int trow = qt*128 + wave*16 + r + 8*hi;
      int col  = head*HD_ + nt*16 + nn;
      float v = Oacc[nt][r] / lrun[r];
      cat[(size_t)trow * CAT_ + col] = f2bf(v);
    }
}

// ---------------------------------------------------------------------------
// Host-side orchestration
// ---------------------------------------------------------------------------
extern "C" void kernel_launch(void* const* d_in, const int* in_sizes, int n_in,
                              void* d_out, int out_size, void* d_ws, size_t ws_size,
                              hipStream_t stream) {
  (void)in_sizes; (void)n_in; (void)out_size; (void)ws_size;
  const float* x       = (const float*)d_in[0];
  const float* vec     = (const float*)d_in[1];
  const float* pe      = (const float*)d_in[2];
  const float* mod_w   = (const float*)d_in[3];
  const float* mod_b   = (const float*)d_in[4];
  const float* lin1_w  = (const float*)d_in[5];
  const float* lin1_b  = (const float*)d_in[6];
  const float* lin2_w  = (const float*)d_in[7];
  const float* lin2_b  = (const float*)d_in[8];
  const float* q_scale = (const float*)d_in[9];
  const float* k_scale = (const float*)d_in[10];
  float* out = (float*)d_out;

  char* ws = (char*)d_ws;
  size_t off = 0;
  auto take = [&](size_t bytes) -> char* {
    char* p = ws + off;
    off += (bytes + 255) & ~(size_t)255;
    return p;
  };
  float*          modbuf = (float*)         take((size_t)3*H_*sizeof(float));
  unsigned short* xmod   = (unsigned short*)take((size_t)L_*H_*2);
  unsigned short* w1b    = (unsigned short*)take((size_t)OUT1_*H_*2);
  unsigned short* w2b    = (unsigned short*)take((size_t)H_*CAT_*2);
  unsigned short* hbuf   = (unsigned short*)take((size_t)L_*OUT1_*2);
  unsigned short* qbuf   = (unsigned short*)take((size_t)NH_*L_*HD_*2);
  unsigned short* kbuf   = (unsigned short*)take((size_t)NH_*L_*HD_*2);
  unsigned short* vbuf   = (unsigned short*)take((size_t)NH_*L_*HD_*2);
  unsigned short* catb   = (unsigned short*)take((size_t)L_*CAT_*2);

  // 1) modulation vector
  mod_gemv_kernel<<<3*H_, 128, 0, stream>>>(vec, mod_w, mod_b, modbuf);
  // 2) LN + modulate -> bf16
  ln_mod_kernel<<<L_, 256, 0, stream>>>(x, modbuf, xmod);
  // 3) weight converts (deterministic each call)
  cvt_bf16_kernel<<<4096, 256, 0, stream>>>(lin1_w, w1b, (long)OUT1_*H_);
  cvt_bf16_kernel<<<4096, 256, 0, stream>>>(lin2_w, w2b, (long)H_*CAT_);
  // 4) GEMM1: h = x_mod @ lin1_w^T + b1  (bf16 out)
  dim3 g1(OUT1_/256, L_/128);
  gemm_bf16_wmma<0><<<g1, 256, 0, stream>>>(xmod, w1b, lin1_b, L_, OUT1_, H_,
                                            hbuf, nullptr, nullptr, nullptr);
  // 5) RMS-norm + RoPE + repack
  qkv_post_kernel<<<L_*NH_, 128, 0, stream>>>(hbuf, pe, q_scale, k_scale,
                                              qbuf, kbuf, vbuf);
  // 6) GELU(mlp) -> concat
  gelu_kernel<<<4096, 256, 0, stream>>>(hbuf, catb);
  // 7) attention -> concat cols [0,H)
  flash_attn_kernel<<<NH_*(L_/128), 256, 0, stream>>>(qbuf, kbuf, vbuf, catb);
  // 8) GEMM2 + residual + gate + clip -> d_out (f32)
  dim3 g2(H_/256, L_/128);
  gemm_bf16_wmma<1><<<g2, 256, 0, stream>>>(catb, w2b, lin2_b, L_, H_, CAT_,
                                            nullptr, x, modbuf, out);
}
